// WhisperAttentionTVM_24962349925053
// MI455X (gfx1250) — matrix operand order repaired and checked by reading, not compiled, not run on hardware
//
#include <hip/hip_runtime.h>

#define SEQ   1500
#define SEQP  1504
#define NH    20
#define HD    64
#define EMB   1280
#define BSZ   4
#define MROWS (BSZ*SEQ)        /* 6000 */
#define MT32  ((MROWS + 31) / 32)  /* 188 */

typedef __bf16 bf16;
typedef __attribute__((ext_vector_type(16))) __bf16 bf16x16;
typedef __attribute__((ext_vector_type(8)))  __bf16 bf16x8;
typedef __attribute__((ext_vector_type(8)))  float  f32x8;
typedef __attribute__((ext_vector_type(4)))  float  f32x4;

static __device__ inline f32x8 wmma_bf16(bf16x16 a, bf16x16 b, f32x8 c) {
  return __builtin_amdgcn_wmma_f32_16x16x32_bf16(false, a, false, b, (short)0, c,
                                                 false, false);
}

// 16 contiguous bf16 (B-operand chunk): element e = mem[e]
static __device__ inline bf16x16 ld_contig16(const bf16* p) {
  union { bf16x16 v; bf16x8 h[2]; } u;
  u.h[0] = *(const bf16x8*)(p);
  u.h[1] = *(const bf16x8*)(p + 8);
  return u.v;
}

// A-operand per-lane chunks: K offsets [16*hh, 16*hh+8) and [16*hh+8, 16*hh+16)
static __device__ inline bf16x16 ld_a_tile(const bf16* p, int hh) {
  union { bf16x16 v; bf16x8 h[2]; } u;
  u.h[0] = *(const bf16x8*)(p + 16 * hh);
  u.h[1] = *(const bf16x8*)(p + 16 * hh + 8);
  return u.v;
}

static __device__ inline unsigned pack2bf(float a, float b) {
  union { bf16 h[2]; unsigned u; } t;
  t.h[0] = (bf16)a; t.h[1] = (bf16)b;
  return t.u;
}

// exchange with lane^16 (SWAPX16: xor=0x10, and=0x1f -> offset 0x401F)
static __device__ inline float swaphalf_f(float x) {
  return __int_as_float(__builtin_amdgcn_ds_swizzle(__float_as_int(x), 0x401F));
}
static __device__ inline unsigned swaphalf_u(unsigned x) {
  return (unsigned)__builtin_amdgcn_ds_swizzle((int)x, 0x401F);
}

static __device__ inline f32x8 zero8() {
  f32x8 z;
#pragma unroll
  for (int i = 0; i < 8; ++i) z[i] = 0.0f;
  return z;
}

// ---------------------------------------------------------------- fp32 -> bf16
__global__ __launch_bounds__(256) void cvt_kernel(const float* __restrict__ s,
                                                  bf16* __restrict__ d, int n8) {
  int i = blockIdx.x * 256 + threadIdx.x;
  if (i >= n8) return;
  const f32x4* sp = (const f32x4*)(s + (size_t)i * 8);
  f32x4 a = sp[0], b = sp[1];
  union { bf16x8 v; bf16 e[8]; } u;
#pragma unroll
  for (int k = 0; k < 4; ++k) { u.e[k] = (bf16)a[k]; u.e[4 + k] = (bf16)b[k]; }
  *(bf16x8*)(d + (size_t)i * 8) = u.v;
}

// ------------------------------------------------ GEMM core: 32M x 64N per wave
// acc[0..3] = M-subtile 0 (rows m0..m0+15), acc[4..7] = subtile 1 (+16 rows),
// double-buffered over K in steps of 32, unrolled 2x (ping-pong).
#define GEMM_CORE(X, W, m0, n0, r, hh, acc)                                     \
  {                                                                             \
    const int row0 = (m0) + (r);                                                \
    const int row1 = ((m0) + 16 + (r) < MROWS) ? ((m0) + 16 + (r)) : (MROWS-1); \
    const bf16* arow0 = (X) + (size_t)row0 * EMB;                               \
    const bf16* arow1 = (X) + (size_t)row1 * EMB;                               \
    const bf16* wbase = (W) + (size_t)((n0) + (r)) * EMB + 16 * (hh);           \
    bf16x16 a0[2], a1[2], b0[4], b1[4];                                         \
    a0[0] = ld_a_tile(arow0, hh);                                               \
    a0[1] = ld_a_tile(arow1, hh);                                               \
    _Pragma("unroll")                                                           \
    for (int j = 0; j < 4; ++j) b0[j] = ld_contig16(wbase + (size_t)j*16*EMB);  \
    for (int kk = 0; kk < EMB; kk += 64) {                                      \
      a1[0] = ld_a_tile(arow0 + kk + 32, hh);                                   \
      a1[1] = ld_a_tile(arow1 + kk + 32, hh);                                   \
      _Pragma("unroll")                                                         \
      for (int j = 0; j < 4; ++j)                                               \
        b1[j] = ld_contig16(wbase + (size_t)j*16*EMB + kk + 32);                \
      _Pragma("unroll")                                                         \
      for (int j = 0; j < 4; ++j) {                                             \
        acc[j]     = wmma_bf16(a0[0], b0[j], acc[j]);                           \
        acc[4 + j] = wmma_bf16(a0[1], b0[j], acc[4 + j]);                       \
      }                                                                         \
      if (kk + 64 < EMB) {                                                      \
        a0[0] = ld_a_tile(arow0 + kk + 64, hh);                                 \
        a0[1] = ld_a_tile(arow1 + kk + 64, hh);                                 \
        _Pragma("unroll")                                                       \
        for (int j = 0; j < 4; ++j)                                             \
          b0[j] = ld_contig16(wbase + (size_t)j*16*EMB + kk + 64);              \
      }                                                                         \
      _Pragma("unroll")                                                         \
      for (int j = 0; j < 4; ++j) {                                             \
        acc[j]     = wmma_bf16(a1[0], b1[j], acc[j]);                           \
        acc[4 + j] = wmma_bf16(a1[1], b1[j], acc[4 + j]);                       \
      }                                                                         \
    }                                                                           \
  }

// ------------------------------------------------- QKV projection (bf16 GEMM)
// y[m][n] = sum_k X[m][k] * W[n][k]  (+bias)*scale
__global__ __launch_bounds__(256) void proj_kernel(
    const bf16* __restrict__ X, const bf16* __restrict__ W,
    const float* __restrict__ bias, float scale, int vtrans,
    bf16* __restrict__ out) {
  const int lane = threadIdx.x & 31;
  const int wid = threadIdx.x >> 5;
  const int r = lane & 15, hh = lane >> 4;
  const int mt = blockIdx.y * 8 + wid;
  if (mt >= MT32) return;
  const int m0 = mt * 32;
  const int n0 = blockIdx.x * 64;

  f32x8 acc[8];
#pragma unroll
  for (int j = 0; j < 8; ++j) acc[j] = zero8();

  GEMM_CORE(X, W, m0, n0, r, hh, acc)

#pragma unroll
  for (int j = 0; j < 4; ++j) {
    const int n = n0 + j * 16 + r;
    const float bb = bias ? bias[n] : 0.0f;
    const int head = n >> 6, d = n & 63;
#pragma unroll
    for (int sub = 0; sub < 2; ++sub)
#pragma unroll
      for (int i = 0; i < 8; ++i) {
        const int m = m0 + sub * 16 + 8 * hh + i;   // C slot i -> row i+8*hh
        if (m < MROWS) {
          const int b = m / SEQ, s = m % SEQ;
          const float v = (acc[sub * 4 + j][i] + bb) * scale;
          const size_t bh = (size_t)b * NH + head;
          if (vtrans) out[(bh * HD + d) * SEQP + s] = (bf16)v;   // V^T [d][s]
          else        out[(bh * SEQP + s) * HD + d] = (bf16)v;   // Q/K [s][d]
        }
      }
  }
}

// ---------------------------------------------------- flash attention per wave
// per wave: 16 queries; loop over 32-key tiles; S^T = K x Q^T so softmax
// reductions are in-lane + one half-swap; O^T = V^T x P^T.
// All loads issued up-front each iteration (K first, then V^T) so the score
// WMMAs wait only on K while V^T latency is hidden behind the softmax VALU.
__global__ __launch_bounds__(256) void attn_kernel(
    const bf16* __restrict__ Q, const bf16* __restrict__ K,
    const bf16* __restrict__ VT, const float* __restrict__ mask,
    bf16* __restrict__ ctx) {
  const int lane = threadIdx.x & 31;
  const int wid = threadIdx.x >> 5;
  const int r = lane & 15, hh = lane >> 4;
  const int q0 = (blockIdx.x * 8 + wid) * 16;
  if (q0 >= SEQP) return;
  const int bh = blockIdx.y;
  const int b = bh / NH, head = bh % NH;
  const int qg = q0 + r;                    // this lane's query column

  // Q^T as persistent B-operand: element e = Q[qg][dbase + 16*hh + e]
  const bf16* qrow = Q + ((size_t)bh * SEQP + qg) * HD;
  const bf16x16 qB0 = ld_contig16(qrow + 16 * hh);        // d 0..31
  const bf16x16 qB1 = ld_contig16(qrow + 32 + 16 * hh);   // d 32..63

  f32x8 O[4];
#pragma unroll
  for (int j = 0; j < 4; ++j) O[j] = zero8();
  float mrun = -1e30f, lrun = 0.0f;
  const float* mrow = mask + ((size_t)b * SEQ + qg) * SEQ;
  const bf16* kbase = K + (size_t)bh * SEQP * HD + (size_t)r * HD;
  const bf16* vbase = VT + (size_t)bh * HD * SEQP + (size_t)r * SEQP;

  for (int k0 = 0; k0 < SEQP; k0 += 32) {
    // ---- issue ALL loads: K tiles first, then V^T tiles
    const bf16* krow0 = kbase + (size_t)k0 * HD;
    bf16x16 ka[4];
    ka[0] = ld_a_tile(krow0, hh);                  // keys t0, d 0..31
    ka[1] = ld_a_tile(krow0 + 32, hh);             // keys t0, d 32..63
    ka[2] = ld_a_tile(krow0 + 16 * HD, hh);        // keys t1, d 0..31
    ka[3] = ld_a_tile(krow0 + 16 * HD + 32, hh);   // keys t1, d 32..63
    bf16x16 va[4];
#pragma unroll
    for (int dt = 0; dt < 4; ++dt)
      va[dt] = ld_a_tile(vbase + (size_t)dt * 16 * SEQP + k0, hh);

    // ---- scores S^T[key][q], two 16-key subtiles
    f32x8 sc[2];
    sc[0] = zero8();
    sc[0] = wmma_bf16(ka[0], qB0, sc[0]);
    sc[0] = wmma_bf16(ka[1], qB1, sc[0]);
    sc[1] = zero8();
    sc[1] = wmma_bf16(ka[2], qB0, sc[1]);
    sc[1] = wmma_bf16(ka[3], qB1, sc[1]);

    // ---- attention mask add + OOB key masking (slot i -> key k0+16t+8hh+i)
    if (k0 + 32 <= SEQ) {
      if (qg < SEQ) {
#pragma unroll
        for (int t = 0; t < 2; ++t) {
          const float2* mp = (const float2*)(mrow + k0 + t * 16 + 8 * hh);
#pragma unroll
          for (int k2 = 0; k2 < 4; ++k2) {
            float2 mv = mp[k2];
            sc[t][2 * k2] += mv.x;
            sc[t][2 * k2 + 1] += mv.y;
          }
        }
      }
    } else {
#pragma unroll
      for (int t = 0; t < 2; ++t)
#pragma unroll
        for (int i = 0; i < 8; ++i) {
          const int kg = k0 + t * 16 + 8 * hh + i;
          if (kg >= SEQ) sc[t][i] = -1e30f;
          else if (qg < SEQ) sc[t][i] += mrow[kg];
        }
    }
    // ---- online softmax (per query = per lane; keys split across halves)
    float mx = -1e30f;
#pragma unroll
    for (int t = 0; t < 2; ++t)
#pragma unroll
      for (int i = 0; i < 8; ++i) mx = fmaxf(mx, sc[t][i]);
    mx = fmaxf(mx, swaphalf_f(mx));
    const float mnew = fmaxf(mrun, mx);
    const float alpha = __expf(mrun - mnew);
    float p[16];
    float rs = 0.0f;
#pragma unroll
    for (int t = 0; t < 2; ++t)
#pragma unroll
      for (int i = 0; i < 8; ++i) {
        const float e = __expf(sc[t][i] - mnew);
        p[t * 8 + i] = e;
        rs += e;
      }
    rs += swaphalf_f(rs);
    lrun = lrun * alpha + rs;
    mrun = mnew;
#pragma unroll
    for (int j = 0; j < 4; ++j)
#pragma unroll
      for (int i = 0; i < 8; ++i) O[j][i] *= alpha;

    // ---- rebuild P^T into B-operand layout with packed half-swaps
    unsigned pk0[4], pk1[4];
#pragma unroll
    for (int j2 = 0; j2 < 4; ++j2) {
      pk0[j2] = pack2bf(p[2 * j2], p[2 * j2 + 1]);          // keys k0+8hh+2j..
      pk1[j2] = pack2bf(p[8 + 2 * j2], p[8 + 2 * j2 + 1]);  // keys k0+16+8hh+2j..
    }
    union { bf16x16 v; unsigned u[8]; } Bp;
#pragma unroll
    for (int j2 = 0; j2 < 4; ++j2) {
      const unsigned x = hh ? pk0[j2] : pk1[j2];  // give other half what it needs
      const unsigned y = swaphalf_u(x);
      Bp.u[j2]     = hh ? y : pk0[j2];            // keys 16hh + 0..7
      Bp.u[4 + j2] = hh ? pk1[j2] : y;            // keys 16hh + 8..15
    }
    // ---- O^T += V^T x P^T  (4 d-tiles of 16) — va already resident
#pragma unroll
    for (int dt = 0; dt < 4; ++dt) O[dt] = wmma_bf16(va[dt], Bp.v, O[dt]);
  }
  // ---- normalize + store ctx[b][s][head*64 + d]  (slot i -> d = 16dt+8hh+i)
  if (qg < SEQ) {
    const float inv = 1.0f / lrun;
#pragma unroll
    for (int dt = 0; dt < 4; ++dt) {
      union { bf16x8 v; bf16 e[8]; } s8;
#pragma unroll
      for (int i = 0; i < 8; ++i) s8.e[i] = (bf16)(O[dt][i] * inv);
      *(bf16x8*)(ctx + ((size_t)b * SEQ + qg) * EMB + head * HD + dt * 16 +
                 8 * hh) = s8.v;
    }
  }
}

// --------------------------------------------------- output projection (fp32)
__global__ __launch_bounds__(256) void outproj_kernel(
    const bf16* __restrict__ X, const bf16* __restrict__ W,
    const float* __restrict__ bias, float* __restrict__ out) {
  const int lane = threadIdx.x & 31;
  const int wid = threadIdx.x >> 5;
  const int r = lane & 15, hh = lane >> 4;
  const int mt = blockIdx.y * 8 + wid;
  if (mt >= MT32) return;
  const int m0 = mt * 32;
  const int n0 = blockIdx.x * 64;

  f32x8 acc[8];
#pragma unroll
  for (int j = 0; j < 8; ++j) acc[j] = zero8();

  GEMM_CORE(X, W, m0, n0, r, hh, acc)

#pragma unroll
  for (int j = 0; j < 4; ++j) {
    const int n = n0 + j * 16 + r;
    const float bb = bias[n];
#pragma unroll
    for (int sub = 0; sub < 2; ++sub)
#pragma unroll
      for (int i = 0; i < 8; ++i) {
        const int m = m0 + sub * 16 + 8 * hh + i;
        if (m < MROWS) out[(size_t)m * EMB + n] = acc[sub * 4 + j][i] + bb;
      }
  }
}

extern "C" void kernel_launch(void* const* d_in, const int* in_sizes, int n_in,
                              void* d_out, int out_size, void* d_ws,
                              size_t ws_size, hipStream_t stream) {
  (void)in_sizes; (void)n_in; (void)out_size; (void)ws_size;
  const float* hidden = (const float*)d_in[0];
  const float* mask = (const float*)d_in[1];
  const float* Wq = (const float*)d_in[2];
  const float* bq = (const float*)d_in[3];
  const float* Wk = (const float*)d_in[4];
  const float* Wv = (const float*)d_in[5];
  const float* bv = (const float*)d_in[6];
  const float* Wo = (const float*)d_in[7];
  const float* bo = (const float*)d_in[8];
  float* out = (float*)d_out;

  bf16* p = (bf16*)d_ws;
  bf16* Xb = p;  p += (size_t)MROWS * EMB;
  bf16* WqB = p; p += (size_t)EMB * EMB;
  bf16* WkB = p; p += (size_t)EMB * EMB;
  bf16* WvB = p; p += (size_t)EMB * EMB;
  bf16* WoB = p; p += (size_t)EMB * EMB;
  bf16* Qb = p;  p += (size_t)BSZ * NH * SEQP * HD;
  bf16* Kb = p;  p += (size_t)BSZ * NH * SEQP * HD;
  bf16* VTb = p; p += (size_t)BSZ * NH * HD * SEQP;
  bf16* Ctx = p; p += (size_t)MROWS * EMB;

  {
    int n8 = MROWS * EMB / 8;
    cvt_kernel<<<(n8 + 255) / 256, 256, 0, stream>>>(hidden, Xb, n8);
  }
  {
    int n8 = EMB * EMB / 8;
    cvt_kernel<<<(n8 + 255) / 256, 256, 0, stream>>>(Wq, WqB, n8);
    cvt_kernel<<<(n8 + 255) / 256, 256, 0, stream>>>(Wk, WkB, n8);
    cvt_kernel<<<(n8 + 255) / 256, 256, 0, stream>>>(Wv, WvB, n8);
    cvt_kernel<<<(n8 + 255) / 256, 256, 0, stream>>>(Wo, WoB, n8);
  }

  dim3 pg(EMB / 64, (MT32 + 7) / 8);
  proj_kernel<<<pg, 256, 0, stream>>>(Xb, WqB, bq, 0.125f, 0, Qb);
  proj_kernel<<<pg, 256, 0, stream>>>(Xb, WkB, nullptr, 1.0f, 0, Kb);
  proj_kernel<<<pg, 256, 0, stream>>>(Xb, WvB, bv, 1.0f, 1, VTb);

  attn_kernel<<<dim3((SEQP / 16 + 7) / 8, BSZ * NH), 256, 0, stream>>>(
      Qb, Kb, VTb, mask, Ctx);

  outproj_kernel<<<pg, 256, 0, stream>>>(Ctx, WoB, bo, out);
}
